// SparseMoETransformer_44736379355522
// MI455X (gfx1250) — matrix operand order repaired and checked by reading, not compiled, not run on hardware
//
#include <hip/hip_runtime.h>
#include <hip/hip_bf16.h>

// ---------------------------------------------------------------------------
// Types / constants
// ---------------------------------------------------------------------------
typedef __attribute__((ext_vector_type(16))) __bf16 v16bf;
typedef __attribute__((ext_vector_type(8)))  float  v8f;

#define TB  256   // threads per block (8 wave32)
#define BMt 64    // M tile
#define BNt 128   // N tile
#define BKt 32    // K tile (== WMMA K for bf16). All GEMM K here are multiples of 32.

__device__ __forceinline__ unsigned short f2bf(float f) {
  unsigned u = __float_as_uint(f);
  u += 0x7FFFu + ((u >> 16) & 1u);   // round-to-nearest-even
  return (unsigned short)(u >> 16);
}
__device__ __forceinline__ unsigned pk2(float lo, float hi) {
  return (unsigned)f2bf(lo) | ((unsigned)f2bf(hi) << 16);
}

struct Frag { union { v16bf v; unsigned u[8]; uint4 u4[2]; }; };

// A-matrix 16x32 bf16 fragment (ISA 7.12.2): per lane the 16 halfs are two
// contiguous 16-byte chunks: halfs [8*half, 8*half+8) and [16+8*half, +8).
// -> two ds_load_b128 per fragment.
__device__ __forceinline__ void load_frag_a(const unsigned short* lds,
                                            int rowBase, int lane, Frag& f) {
  int r = lane & 15, hs = lane >> 4;
  const unsigned short* p = lds + (rowBase + r) * BKt + 8 * hs;
  f.u4[0] = *(const uint4*)(p);        // K = 8*hs .. 8*hs+7
  f.u4[1] = *(const uint4*)(p + 16);   // K = 16+8*hs .. 16+8*hs+7
}

// B-matrix 32x16 bf16 fragment: same K order as A; with the LDS tile stored
// [n][k] the 16 halfs are K [8*half, +8) and [16+8*half, +8).
// -> two ds_load_b128 per fragment.
__device__ __forceinline__ void load_frag_b(const unsigned short* lds,
                                            int rowBase, int lane, Frag& f) {
  int r = lane & 15, hs = lane >> 4;
  const unsigned short* p = lds + (rowBase + r) * BKt + 8 * hs;
  f.u4[0] = *(const uint4*)(p);
  f.u4[1] = *(const uint4*)(p + 16);
}

__device__ __forceinline__ v8f wmma_bf16(const Frag& a, const Frag& b, v8f c) {
  return __builtin_amdgcn_wmma_f32_16x16x32_bf16(
      false, a.v, false, b.v, (short)0, c, false, false);
}

// ---------------------------------------------------------------------------
// Workhorse GEMM:  C[z] = op(A[z] x B[z] + bias[z])
//   BT:      B given as [N,K] row-major (i.e. multiply by B^T)
//   RELU:    relu epilogue
//   GATHER:  A rows indirected through rowIdx (token routing)
//   SCATTER: C rows indirected through rowIdx, gate-scaled, atomic-accum
//   dM:      optional per-z M override (device counts) for routed experts
// z decomposed as (z/zDiv, z%zDiv) with separate strides per operand.
// Requires: K % 32 == 0 (true for all call sites: 64/512/768/3072).
// ---------------------------------------------------------------------------
template <bool BT, bool RELU, bool GATHER, bool SCATTER>
__global__ __launch_bounds__(TB) void gemm_wmma(
    const float* __restrict__ A, long long sAo, long long sAi, int lda,
    const float* __restrict__ Bm, long long sBo, long long sBi, int ldb,
    const float* __restrict__ bias, long long sBiasO, long long sBiasI,
    float* __restrict__ C, long long sCo, long long sCi, int ldc,
    int M, int N, int K, int zDiv,
    const int* __restrict__ dM,
    const int* __restrict__ rowIdx, int rowIdxStride,
    const float* __restrict__ rowScale, int rowScaleStride) {
  const int z  = blockIdx.z;
  const int zo = z / zDiv, zi = z % zDiv;
  A  += zo * sAo + zi * sAi;
  Bm += zo * sBo + zi * sBi;
  if (bias) bias += zo * sBiasO + zi * sBiasI;
  C  += zo * sCo + zi * sCi;
  const int*   rIdx = rowIdx   ? rowIdx   + (long long)z * rowIdxStride   : nullptr;
  const float* rScl = rowScale ? rowScale + (long long)z * rowScaleStride : nullptr;

  const int Me    = dM ? dM[z] : M;
  const int mBase = blockIdx.y * BMt;
  const int nBase = blockIdx.x * BNt;
  if (mBase >= Me) return;  // uniform per block

  __shared__ __align__(16) unsigned short sA[2][BMt * BKt];
  __shared__ __align__(16) unsigned short sB[2][BNt * BKt];

  const int tid  = threadIdx.x;
  const int lane = tid & 31;
  const int w    = tid >> 5;
  const int wm   = w >> 2;   // 0..1
  const int wn   = w & 3;    // 0..3

  // --- A staging coords: 64x32 tile, 8 contiguous K per thread ---
  const int a_m = (tid * 8) >> 5;
  const int a_k = (tid * 8) & 31;
  const int a_grow = mBase + a_m;
  const bool a_val = a_grow < Me;
  long long a_row;
  if (GATHER) a_row = a_val ? rIdx[a_grow] : 0;
  else        a_row = a_val ? a_grow : 0;
  const float* aP = A + a_row * (long long)lda + a_k;

  // --- B staging coords ---
  // BT path:   thread covers 16 contiguous K of one n (two threads per n).
  // non-BT:    thread covers an 8K x 2N patch (kg = K-octet, n2 = even col),
  //            so LDS writes are uint4 (ds_store_b128) in both paths.
  int b_n = 0, b_k = 0, b_kg = 0, b_n2 = 0;
  bool b_v0 = true, b_v1 = true;
  const float* bP = Bm;
  if (BT) {
    b_n = (tid * 16) >> 5;        // 0..127
    b_k = (tid * 16) & 31;        // 0 or 16
    int gn = nBase + b_n;
    b_v0 = gn < N;
    bP = Bm + (long long)(b_v0 ? gn : 0) * ldb + b_k;
  } else {
    b_kg = tid >> 6;              // 0..3  -> K rows [8*kg, 8*kg+8)
    b_n2 = (tid & 63) * 2;        // 0..126
    int c0 = nBase + b_n2;
    b_v0 = c0 < N;
    b_v1 = (c0 + 1) < N;
  }

  v8f c00 = {}, c01 = {}, c10 = {}, c11 = {};
  const int nk = K / BKt;

  auto stage = [&](int buf, int k0) {
    // ---- A: two b128 global loads, one b128 LDS store ----
    if (k0 + BKt < K) __builtin_prefetch(aP + k0 + BKt, 0, 1);
    float fa[8];
    #pragma unroll
    for (int i = 0; i < 8; ++i) {
      float t = aP[k0 + i];
      fa[i] = a_val ? t : 0.f;
    }
    uint4 pa;
    pa.x = pk2(fa[0], fa[1]); pa.y = pk2(fa[2], fa[3]);
    pa.z = pk2(fa[4], fa[5]); pa.w = pk2(fa[6], fa[7]);
    *(uint4*)&sA[buf][a_m * BKt + a_k] = pa;

    // ---- B ----
    if (BT) {
      float fb[16];
      #pragma unroll
      for (int i = 0; i < 16; ++i) {
        float t = bP[k0 + i];
        fb[i] = b_v0 ? t : 0.f;
      }
      uint4 p0, p1;
      p0.x = pk2(fb[0], fb[1]);   p0.y = pk2(fb[2], fb[3]);
      p0.z = pk2(fb[4], fb[5]);   p0.w = pk2(fb[6], fb[7]);
      p1.x = pk2(fb[8], fb[9]);   p1.y = pk2(fb[10], fb[11]);
      p1.z = pk2(fb[12], fb[13]); p1.w = pk2(fb[14], fb[15]);
      *(uint4*)&sB[buf][b_n * BKt + b_k] = p0;
      *(uint4*)&sB[buf][b_n * BKt + b_k + 8] = p1;
    } else {
      int c0 = nBase + b_n2;
      int cc0 = b_v0 ? c0 : 0;
      int cc1 = b_v1 ? (c0 + 1) : 0;
      float f0[8], f1[8];
      #pragma unroll
      for (int kk = 0; kk < 8; ++kk) {
        const float* rrow = Bm + (long long)(k0 + b_kg * 8 + kk) * ldb;
        float t0 = rrow[cc0];
        float t1 = rrow[cc1];
        f0[kk] = b_v0 ? t0 : 0.f;
        f1[kk] = b_v1 ? t1 : 0.f;
      }
      uint4 p0, p1;
      p0.x = pk2(f0[0], f0[1]); p0.y = pk2(f0[2], f0[3]);
      p0.z = pk2(f0[4], f0[5]); p0.w = pk2(f0[6], f0[7]);
      p1.x = pk2(f1[0], f1[1]); p1.y = pk2(f1[2], f1[3]);
      p1.z = pk2(f1[4], f1[5]); p1.w = pk2(f1[6], f1[7]);
      *(uint4*)&sB[buf][b_n2 * BKt + b_kg * 8] = p0;
      *(uint4*)&sB[buf][(b_n2 + 1) * BKt + b_kg * 8] = p1;
    }
  };

  stage(0, 0);
  __syncthreads();
  for (int kt = 0; kt < nk; ++kt) {
    const int cur = kt & 1;
    if (kt + 1 < nk) stage(cur ^ 1, (kt + 1) * BKt);
    const unsigned short* aL = sA[cur];
    const unsigned short* bL = sB[cur];
    Frag a0, a1, b0, b1;
    load_frag_a(aL, wm * 32,      lane, a0);
    load_frag_a(aL, wm * 32 + 16, lane, a1);
    load_frag_b(bL, wn * 32,      lane, b0);
    load_frag_b(bL, wn * 32 + 16, lane, b1);
    c00 = wmma_bf16(a0, b0, c00);
    c01 = wmma_bf16(a0, b1, c01);
    c10 = wmma_bf16(a1, b0, c10);
    c11 = wmma_bf16(a1, b1, c11);
    __syncthreads();
  }

  // --- epilogue: C/D 16x16 f32 layout: reg r -> M = 8*(lane/16)+r, N = lane%16
  const int wmBase = mBase + wm * 32;
  const int wnBase = nBase + wn * 32;
  const int nOff = lane & 15;
  const int mOff = (lane >> 4) * 8;
  #pragma unroll
  for (int ti = 0; ti < 2; ++ti) {
    #pragma unroll
    for (int tj = 0; tj < 2; ++tj) {
      v8f acc = (ti == 0) ? (tj == 0 ? c00 : c01) : (tj == 0 ? c10 : c11);
      const int mb = wmBase + ti * 16;
      const int nb = wnBase + tj * 16;
      #pragma unroll
      for (int r = 0; r < 8; ++r) {
        int row = mb + mOff + r;
        int col = nb + nOff;
        if (row < Me && col < N) {
          float v = acc[r];
          if (bias) v += bias[col];
          if (SCATTER) {
            float g = rScl ? rScl[row] : 1.f;
            long long orow = rIdx[row];
            atomicAdd(&C[orow * (long long)ldc + col], v * g);
          } else {
            if (RELU) v = fmaxf(v, 0.f);
            C[(long long)row * ldc + col] = v;
          }
        }
      }
    }
  }
}

// ---------------------------------------------------------------------------
// Support kernels
// ---------------------------------------------------------------------------
__global__ void embed_kernel(const int* __restrict__ toks,
                             const float* __restrict__ temb,
                             const float* __restrict__ pemb,
                             float* __restrict__ x, int S, int E) {
  int row = blockIdx.x;
  int s = row % S;
  int tok = toks[row];
  const float* te = temb + (long long)tok * E;
  const float* pe = pemb + (long long)s * E;
  float* xr = x + (long long)row * E;
  for (int i = threadIdx.x; i < E; i += blockDim.x) xr[i] = te[i] + pe[i];
}

__global__ void ln_kernel(const float* __restrict__ x, const float* __restrict__ g,
                          const float* __restrict__ b, float* __restrict__ y, int n) {
  __shared__ float red[TB];
  int row = blockIdx.x, tid = threadIdx.x;
  const float* xr = x + (long long)row * n;
  float* yr = y + (long long)row * n;
  float s = 0.f;
  for (int i = tid; i < n; i += TB) s += xr[i];
  red[tid] = s; __syncthreads();
  for (int o = TB / 2; o; o >>= 1) { if (tid < o) red[tid] += red[tid + o]; __syncthreads(); }
  float mean = red[0] / n;
  __syncthreads();
  float v = 0.f;
  for (int i = tid; i < n; i += TB) { float d = xr[i] - mean; v += d * d; }
  red[tid] = v; __syncthreads();
  for (int o = TB / 2; o; o >>= 1) { if (tid < o) red[tid] += red[tid + o]; __syncthreads(); }
  float inv = rsqrtf(red[0] / n + 1e-5f);
  for (int i = tid; i < n; i += TB) yr[i] = (xr[i] - mean) * inv * g[i] + b[i];
}

__global__ void softmax_causal(float* __restrict__ p, int S, float scale) {
  __shared__ float red[TB];
  long long row = blockIdx.x;
  int q = (int)(row % S);
  float* pr = p + row * (long long)S;
  int tid = threadIdx.x;
  int valid = q + 1;
  float mx = -3.4e38f;
  for (int i = tid; i < valid; i += TB) mx = fmaxf(mx, pr[i] * scale);
  red[tid] = mx; __syncthreads();
  for (int o = TB / 2; o; o >>= 1) { if (tid < o) red[tid] = fmaxf(red[tid], red[tid + o]); __syncthreads(); }
  float m = red[0];
  __syncthreads();
  float s = 0.f;
  for (int i = tid; i < valid; i += TB) { float e = __expf(pr[i] * scale - m); pr[i] = e; s += e; }
  red[tid] = s; __syncthreads();
  for (int o = TB / 2; o; o >>= 1) { if (tid < o) red[tid] += red[tid + o]; __syncthreads(); }
  float inv = 1.f / red[0];
  for (int i = tid; i < S; i += TB) pr[i] = (i < valid) ? pr[i] * inv : 0.f;
}

__global__ void topk_gates(const float* __restrict__ rs, float* __restrict__ g,
                           int n, int NE) {
  int t = blockIdx.x * blockDim.x + threadIdx.x;
  if (t >= n) return;
  const float* r = rs + (long long)t * NE;
  float b1 = -3.4e38f, b2 = -3.4e38f;
  int i1 = 0, i2 = 0;
  for (int i = 0; i < NE; ++i) {
    float v = r[i];
    if (v > b1)      { b2 = b1; i2 = i1; b1 = v; i1 = i; }
    else if (v > b2) { b2 = v; i2 = i; }
  }
  float e2 = __expf(b2 - b1);
  float inv = 1.f / (1.f + e2);
  float* gr = g + (long long)t * NE;
  for (int i = 0; i < NE; ++i) gr[i] = 0.f;
  gr[i1] = inv;
  gr[i2] = e2 * inv;
}

__global__ void zero_ints(int* __restrict__ p, int n) {
  int i = blockIdx.x * blockDim.x + threadIdx.x;
  if (i < n) p[i] = 0;
}

__global__ void route_build(const float* __restrict__ g, int* __restrict__ counts,
                            int* __restrict__ ridx, float* __restrict__ rscale,
                            int n, int NE, int cap) {
  int t = blockIdx.x * blockDim.x + threadIdx.x;
  if (t >= n) return;
  const float* gr = g + (long long)t * NE;
  for (int e = 0; e < NE; ++e) {
    float ge = gr[e];
    if (ge > 0.f) {
      int pos = atomicAdd(&counts[e], 1);
      ridx[e * cap + pos] = t;
      rscale[e * cap + pos] = ge;
    }
  }
}

__global__ void add_kernel(const float* __restrict__ a, const float* __restrict__ b,
                           float* __restrict__ y, int n) {
  int i = blockIdx.x * blockDim.x + threadIdx.x;
  if (i < n) y[i] = a[i] + b[i];
}

__global__ void copy_kernel(const float* __restrict__ a, float* __restrict__ y, int n) {
  int i = blockIdx.x * blockDim.x + threadIdx.x;
  if (i < n) y[i] = a[i];
}

// ---------------------------------------------------------------------------
// Host orchestration
// ---------------------------------------------------------------------------
extern "C" void kernel_launch(void* const* d_in, const int* in_sizes, int n_in,
                              void* d_out, int out_size, void* d_ws, size_t ws_size,
                              hipStream_t stream) {
  (void)in_sizes; (void)n_in; (void)out_size; (void)ws_size;

  const int Bz = 2, S = 512, E = 768, H = 12, NE = 8, L = 2, V = 32000;
  const int HD = 64, FF = 3072;
  const int BS = Bz * S;   // 1024 token rows
  const int BH = Bz * H;   // 24 attention batches

  const int*   tokens    = (const int*)  d_in[0];
  const float* token_emb = (const float*)d_in[1];
  const float* pos_emb   = (const float*)d_in[2];
  const float* ln1_g     = (const float*)d_in[3];
  const float* ln1_b     = (const float*)d_in[4];
  const float* wq        = (const float*)d_in[5];
  const float* bq        = (const float*)d_in[6];
  const float* wk        = (const float*)d_in[7];
  const float* bk        = (const float*)d_in[8];
  const float* wv        = (const float*)d_in[9];
  const float* bv        = (const float*)d_in[10];
  const float* wo        = (const float*)d_in[11];
  const float* bo        = (const float*)d_in[12];
  const float* ln2_g     = (const float*)d_in[13];
  const float* ln2_b     = (const float*)d_in[14];
  const float* rw1       = (const float*)d_in[15];
  const float* rb1       = (const float*)d_in[16];
  const float* rw2       = (const float*)d_in[17];
  const float* rb2       = (const float*)d_in[18];
  const float* ew1       = (const float*)d_in[19];
  const float* eb1       = (const float*)d_in[20];
  const float* ew2       = (const float*)d_in[21];
  const float* eb2       = (const float*)d_in[22];
  const float* lnf_g     = (const float*)d_in[23];
  const float* lnf_b     = (const float*)d_in[24];
  const float* out_w     = (const float*)d_in[25];
  const float* out_b     = (const float*)d_in[26];
  float* logits          = (float*)d_out;

  // --- workspace carve (floats) ---
  float* w = (float*)d_ws;
  size_t off = 0;
  auto alloc = [&](size_t n) { float* p = w + off; off += n; return p; };
  const size_t NBSE = (size_t)BS * E;
  float* bx   = alloc(NBSE);            // residual stream
  float* bh   = alloc(NBSE);            // post-LN1 / final-LN
  float* bh2  = alloc(NBSE);            // post-LN2
  float* bt   = alloc(NBSE);            // temp (proj out / router mid)
  float* bq_  = alloc(NBSE);            // q [B,H,S,HD]
  float* bk_  = alloc(NBSE);            // k
  float* bv_  = alloc(NBSE);            // v
  float* bo_  = alloc(NBSE);            // attn concat [B,S,E]
  float* bs_  = alloc((size_t)BH * S * S);   // scores [B,H,S,S]
  float* brs  = alloc((size_t)BS * NE);      // router logits
  float* bg   = alloc((size_t)BS * NE);      // gates
  float* bscale = alloc((size_t)NE * BS);    // gathered gate per slot
  int* counts = (int*)(w + off); off += 8;
  int* ridx   = (int*)(w + off); off += (size_t)NE * BS;
  float* bmid = alloc((size_t)NE * BS * FF); // expert hidden (capacity BS/expert)

  const dim3 blk(TB);
  const int EW = (int)(NBSE);  // elementwise size

  embed_kernel<<<BS, blk, 0, stream>>>(tokens, token_emb, pos_emb, bx, S, E);

  for (int l = 0; l < L; ++l) {
    const float* l_ln1g = ln1_g + l * E;
    const float* l_ln1b = ln1_b + l * E;
    const float* l_wq = wq + (size_t)l * H * E * HD;
    const float* l_bq = bq + (size_t)l * H * HD;
    const float* l_wk = wk + (size_t)l * H * E * HD;
    const float* l_bk = bk + (size_t)l * H * HD;
    const float* l_wv = wv + (size_t)l * H * E * HD;
    const float* l_bv = bv + (size_t)l * H * HD;
    const float* l_wo = wo + (size_t)l * E * E;
    const float* l_bo = bo + (size_t)l * E;
    const float* l_ln2g = ln2_g + l * E;
    const float* l_ln2b = ln2_b + l * E;
    const float* l_rw1 = rw1 + (size_t)l * E * E;
    const float* l_rb1 = rb1 + (size_t)l * E;
    const float* l_rw2 = rw2 + (size_t)l * E * NE;
    const float* l_rb2 = rb2 + (size_t)l * NE;
    const float* l_ew1 = ew1 + (size_t)l * NE * E * FF;
    const float* l_eb1 = eb1 + (size_t)l * NE * FF;
    const float* l_ew2 = ew2 + (size_t)l * NE * FF * E;
    const float* l_eb2 = eb2 + (size_t)l * NE * E;

    // h = LN1(x)
    ln_kernel<<<BS, blk, 0, stream>>>(bx, l_ln1g, l_ln1b, bh, E);

    // q/k/v: per (b,h): [S,E] x [E,HD] + bias  -> [B,H,S,HD]
    const dim3 gQKV(1, S / BMt, BH);
    gemm_wmma<false, false, false, false><<<gQKV, blk, 0, stream>>>(
        bh, (long long)S * E, 0, E,
        l_wq, 0, (long long)E * HD, HD,
        l_bq, 0, HD,
        bq_, (long long)H * S * HD, (long long)S * HD, HD,
        S, HD, E, H, nullptr, nullptr, 0, nullptr, 0);
    gemm_wmma<false, false, false, false><<<gQKV, blk, 0, stream>>>(
        bh, (long long)S * E, 0, E,
        l_wk, 0, (long long)E * HD, HD,
        l_bk, 0, HD,
        bk_, (long long)H * S * HD, (long long)S * HD, HD,
        S, HD, E, H, nullptr, nullptr, 0, nullptr, 0);
    gemm_wmma<false, false, false, false><<<gQKV, blk, 0, stream>>>(
        bh, (long long)S * E, 0, E,
        l_wv, 0, (long long)E * HD, HD,
        l_bv, 0, HD,
        bv_, (long long)H * S * HD, (long long)S * HD, HD,
        S, HD, E, H, nullptr, nullptr, 0, nullptr, 0);

    // scores = q k^T  (B transposed path)
    gemm_wmma<true, false, false, false><<<dim3(S / BNt, S / BMt, BH), blk, 0, stream>>>(
        bq_, (long long)H * S * HD, (long long)S * HD, HD,
        bk_, (long long)H * S * HD, (long long)S * HD, HD,
        nullptr, 0, 0,
        bs_, (long long)H * S * S, (long long)S * S, S,
        S, S, HD, H, nullptr, nullptr, 0, nullptr, 0);

    softmax_causal<<<BH * S, blk, 0, stream>>>(bs_, S, 0.125f);

    // o = P v  -> concat heads into [B,S,E]
    gemm_wmma<false, false, false, false><<<dim3(1, S / BMt, BH), blk, 0, stream>>>(
        bs_, (long long)H * S * S, (long long)S * S, S,
        bv_, (long long)H * S * HD, (long long)S * HD, HD,
        nullptr, 0, 0,
        bo_, (long long)S * E, (long long)HD, E,
        S, HD, S, H, nullptr, nullptr, 0, nullptr, 0);

    // x = h + (o @ wo + bo)
    gemm_wmma<false, false, false, false><<<dim3(E / BNt, BS / BMt, 1), blk, 0, stream>>>(
        bo_, 0, 0, E,
        l_wo, 0, 0, E,
        l_bo, 0, 0,
        bt, 0, 0, E,
        BS, E, E, 1, nullptr, nullptr, 0, nullptr, 0);
    add_kernel<<<(EW + TB - 1) / TB, blk, 0, stream>>>(bh, bt, bx, EW);

    // h2 = LN2(x)
    ln_kernel<<<BS, blk, 0, stream>>>(bx, l_ln2g, l_ln2b, bh2, E);

    // router: rs = relu(h2 @ rw1 + rb1) @ rw2 + rb2
    gemm_wmma<false, true, false, false><<<dim3(E / BNt, BS / BMt, 1), blk, 0, stream>>>(
        bh2, 0, 0, E,
        l_rw1, 0, 0, E,
        l_rb1, 0, 0,
        bt, 0, 0, E,
        BS, E, E, 1, nullptr, nullptr, 0, nullptr, 0);
    gemm_wmma<false, false, false, false><<<dim3(1, BS / BMt, 1), blk, 0, stream>>>(
        bt, 0, 0, E,
        l_rw2, 0, 0, NE,
        l_rb2, 0, 0,
        brs, 0, 0, NE,
        BS, NE, E, 1, nullptr, nullptr, 0, nullptr, 0);

    topk_gates<<<(BS + TB - 1) / TB, blk, 0, stream>>>(brs, bg, BS, NE);
    zero_ints<<<1, 32, 0, stream>>>(counts, NE);
    route_build<<<(BS + TB - 1) / TB, blk, 0, stream>>>(bg, counts, ridx, bscale,
                                                        BS, NE, BS);

    // residual base: x = h2; experts atomically accumulate gate*eo
    copy_kernel<<<(EW + TB - 1) / TB, blk, 0, stream>>>(bh2, bx, EW);

    // expert up-proj: mid[e] = relu(gather(h2) @ ew1[e] + eb1[e])
    gemm_wmma<false, true, true, false><<<dim3(FF / BNt, BS / BMt, NE), blk, 0, stream>>>(
        bh2, 0, 0, E,
        l_ew1, (long long)E * FF, 0, FF,
        l_eb1, (long long)FF, 0,
        bmid, (long long)BS * FF, 0, FF,
        BS, FF, E, 1,
        counts, ridx, BS, nullptr, 0);

    // expert down-proj: x[token] += gate * (mid[e] @ ew2[e] + eb2[e])
    gemm_wmma<false, false, false, true><<<dim3(E / BNt, BS / BMt, NE), blk, 0, stream>>>(
        bmid, (long long)BS * FF, 0, FF,
        l_ew2, (long long)FF * E, 0, E,
        l_eb2, (long long)E, 0,
        bx, 0, 0, E,
        BS, E, FF, 1,
        counts, ridx, BS, bscale, BS);
  }

  // final LN + logits
  ln_kernel<<<BS, blk, 0, stream>>>(bx, lnf_g, lnf_b, bh, E);
  gemm_wmma<false, false, false, false><<<dim3(V / BNt, BS / BMt, 1), blk, 0, stream>>>(
      bh, 0, 0, E,
      out_w, 0, 0, V,
      out_b, 0, 0,
      logits, 0, 0, V,
      BS, V, E, 1, nullptr, nullptr, 0, nullptr, 0);
}
